// PatchAugNet_66056597012938
// MI455X (gfx1250) — compile-verified, matrix operand order checked by reading, not run on hardware
//
#include <hip/hip_runtime.h>
#include <hip/hip_bf16.h>
#include <stdint.h>

typedef __attribute__((ext_vector_type(16))) _Float16 v16h;
typedef __attribute__((ext_vector_type(8)))  float    v8f;

#define WS_ALIGN(x) (((x) + (size_t)255) & ~(size_t)255)

// ---------------------------------------------------------------------------
// Swizzled WMMA-fragment addressing.
// A 16x32 (or 32x16) f16 operand tile is stored as 512 contiguous halves:
// lane l's v16h fragment lives at (tile*512 + l*16). Element (c=k, m=row/col):
//   ct = m>>4, lo = m&15, kb = c>>5, kr = c&31,
//   hi = (kr>>3)&1, e = (kr&7) + (kr>=16 ? 8 : 0), lane = hi*16+lo
// matching the ISA 16-bit A-matrix layout: k = hi*8 + (e<8 ? e : e+8).
// ---------------------------------------------------------------------------
__device__ __forceinline__ size_t swz_off(int c, int m, int nkb) {
  int ct = m >> 4, lo = m & 15;
  int kb = c >> 5, kr = c & 31;
  int hi = (kr >> 3) & 1;
  int e  = (kr & 7) | ((kr & 16) >> 1);
  return ((size_t)ct * nkb + kb) * 512 + (size_t)((hi << 4) | lo) * 16 + e;
}

// ---------------------------------------------------------------------------
// Farthest point sampling: one block per batch, LDS argmax reduction.
// ---------------------------------------------------------------------------
__global__ void fps_kernel(const float* __restrict__ xyz, int* __restrict__ idx,
                           float* __restrict__ mind, int N, int npoint) {
  int b = blockIdx.x;
  const float* p = xyz + (size_t)b * N * 3;
  float* md = mind + (size_t)b * N;
  int* id = idx + (size_t)b * npoint;
  int tid = threadIdx.x;
  int nt = blockDim.x;
  __shared__ float s_val[256];
  __shared__ int s_idx[256];
  __shared__ int s_last;
  for (int i = tid; i < N; i += nt) md[i] = 1.0e10f;
  if (tid == 0) { s_last = 0; id[0] = 0; }
  __syncthreads();
  for (int it = 1; it < npoint; ++it) {
    int last = s_last;
    float lx = p[last*3+0], ly = p[last*3+1], lz = p[last*3+2];
    float best = -1.0f; int bi = 0;
    for (int i = tid; i < N; i += nt) {
      float dx = p[i*3+0]-lx, dy = p[i*3+1]-ly, dz = p[i*3+2]-lz;
      float d = dx*dx + dy*dy + dz*dz;
      float m = fminf(md[i], d);
      md[i] = m;
      if (m > best) { best = m; bi = i; }
    }
    s_val[tid] = best; s_idx[tid] = bi;
    __syncthreads();
    for (int off = nt >> 1; off > 0; off >>= 1) {
      if (tid < off && s_val[tid+off] > s_val[tid]) {
        s_val[tid] = s_val[tid+off]; s_idx[tid] = s_idx[tid+off];
      }
      __syncthreads();
    }
    if (tid == 0) { id[it] = s_idx[0]; s_last = s_idx[0]; }
    __syncthreads();
  }
}

__global__ void gather_xyz(const float* __restrict__ xyz, const int* __restrict__ cidx,
                           float* __restrict__ nxyz, int N, int Np, int total) {
  int t = blockIdx.x * blockDim.x + threadIdx.x;
  if (t >= total) return;
  int b = t / Np;
  int pi = cidx[t];
  const float* src = xyz + ((size_t)b * N + pi) * 3;
  nxyz[(size_t)t*3+0] = src[0];
  nxyz[(size_t)t*3+1] = src[1];
  nxyz[(size_t)t*3+2] = src[2];
}

// ---------------------------------------------------------------------------
// kNN with dilation 2: keep sorted top-40, emit every 2nd (20 neighbors).
// ---------------------------------------------------------------------------
__global__ void knn_kernel(const float* __restrict__ xyz, const float* __restrict__ nxyz,
                           int* __restrict__ nidx, int N, int Np, int total) {
  const int K2 = 40, S = 20;
  int t = blockIdx.x * blockDim.x + threadIdx.x;
  if (t >= total) return;
  int b = t / Np;
  const float* p = xyz + (size_t)b * N * 3;
  float qx = nxyz[(size_t)t*3+0], qy = nxyz[(size_t)t*3+1], qz = nxyz[(size_t)t*3+2];
  float dl[K2]; int il[K2];
  for (int k = 0; k < K2; ++k) { dl[k] = 3.0e38f; il[k] = 0; }
  for (int i = 0; i < N; ++i) {
    float dx = p[i*3+0]-qx, dy = p[i*3+1]-qy, dz = p[i*3+2]-qz;
    float d = dx*dx + dy*dy + dz*dz;
    if (d < dl[K2-1]) {
      int pos = K2 - 1;
      while (pos > 0 && dl[pos-1] > d) { dl[pos] = dl[pos-1]; il[pos] = il[pos-1]; --pos; }
      dl[pos] = d; il[pos] = i;
    }
  }
  for (int s = 0; s < S; ++s) nidx[(size_t)t * S + s] = il[2*s];
}

// ---------------------------------------------------------------------------
// Build grouped MLP input, K-padded + swizzled: Xsw holds Cin32 x M (f16).
// feats layout: channel-major f16 [Cf][B*N]; feats==nullptr -> use xyz.
// ---------------------------------------------------------------------------
__global__ void group_kernel(const float* __restrict__ xyz, const float* __restrict__ nxyz,
                             const int* __restrict__ nidx, const _Float16* __restrict__ feats,
                             _Float16* __restrict__ Xsw, int N, int Np, int S, int Cf,
                             int B, int Cin32, int total) {
  int t = blockIdx.x * blockDim.x + threadIdx.x;
  if (t >= total) return;
  int j = (t / S) % Np;
  int b = t / (S * Np);
  int nkb = Cin32 >> 5;
  int pi = nidx[t];
  const float* pt = xyz + ((size_t)b * N + pi) * 3;
  const float* ct = nxyz + ((size_t)b * Np + j) * 3;
  Xsw[swz_off(0, t, nkb)] = (_Float16)(pt[0] - ct[0]);
  Xsw[swz_off(1, t, nkb)] = (_Float16)(pt[1] - ct[1]);
  Xsw[swz_off(2, t, nkb)] = (_Float16)(pt[2] - ct[2]);
  if (feats) {
    for (int c = 0; c < Cf; ++c)
      Xsw[swz_off(3 + c, t, nkb)] =
          feats[(size_t)c*((size_t)B*N) + (size_t)b*N + pi];
  } else {
    for (int c = 0; c < 3; ++c)
      Xsw[swz_off(3 + c, t, nkb)] = (_Float16)pt[c];
  }
  for (int c = 3 + Cf; c < Cin32; ++c)   // zero K-padding
    Xsw[swz_off(c, t, nkb)] = (_Float16)0.0f;
}

// ---------------------------------------------------------------------------
// WMMA GEMM on pre-swizzled operands: Y[Cout][M] = A(Cout x K32) * X(K32 x M).
// One wave computes RT row tiles x one 16-col tile: the shared B fragment is
// loaded once per K-step and reused by RT v_wmma ops (X re-read / RT).
// ---------------------------------------------------------------------------
template <int RT>
__global__ void __launch_bounds__(32)
wmma_gemm_rt(const _Float16* __restrict__ Aw, const _Float16* __restrict__ Xw,
             float* __restrict__ Y, int nkb, int M) {
  int tilesM = M >> 4;
  int tile = blockIdx.x;
  int tm = tile % tilesM;
  int trb = tile / tilesM;          // row-block index (RT row tiles)
  int lane = threadIdx.x & 31;
  const v16h* Bp = (const v16h*)Xw + (size_t)tm * nkb * 32 + lane;
  const v16h* Ap = (const v16h*)Aw + (size_t)(trb * RT) * nkb * 32 + lane;
  v8f acc[RT];
#pragma unroll
  for (int r = 0; r < RT; ++r) acc[r] = (v8f){};
#pragma unroll 2
  for (int kb = 0; kb < nkb; ++kb) {
    v16h b = Bp[(size_t)kb * 32];
#pragma unroll
    for (int r = 0; r < RT; ++r) {
      v16h a = Ap[((size_t)r * nkb + kb) * 32];
      acc[r] = __builtin_amdgcn_wmma_f32_16x16x32_f16(false, a, false, b,
                                                      (short)0, acc[r], false, false);
    }
  }
  int hi = lane >> 4, lo = lane & 15;
#pragma unroll
  for (int r = 0; r < RT; ++r) {
#pragma unroll
    for (int v = 0; v < 8; ++v) {
      int row = (trb * RT + r) * 16 + v + 8 * hi;  // C/D layout: VGPR v -> row v+8*(lane/16)
      Y[(size_t)row * M + tm * 16 + lo] = acc[r][v];
    }
  }
}

// ---------------------------------------------------------------------------
// Per-channel batchnorm stats over M -> fused scale/shift.
// ---------------------------------------------------------------------------
__global__ void bn_stats_kernel(const float* __restrict__ Y, const float* __restrict__ gamma,
                                const float* __restrict__ beta, float* __restrict__ scale,
                                float* __restrict__ shift, int M) {
  int c = blockIdx.x;
  const float* row = Y + (size_t)c * M;
  float s = 0.f, s2 = 0.f;
  for (int i = threadIdx.x; i < M; i += blockDim.x) { float v = row[i]; s += v; s2 += v*v; }
  __shared__ float ss[256], sq[256];
  int tid = threadIdx.x;
  ss[tid] = s; sq[tid] = s2;
  __syncthreads();
  for (int off = blockDim.x >> 1; off > 0; off >>= 1) {
    if (tid < off) { ss[tid] += ss[tid+off]; sq[tid] += sq[tid+off]; }
    __syncthreads();
  }
  if (tid == 0) {
    float invM = 1.0f / (float)M;
    float m = ss[0] * invM;
    float var = fmaxf(sq[0] * invM - m * m, 0.0f);
    float a = gamma[c] * rsqrtf(var + 1e-5f);
    scale[c] = a;
    shift[c] = beta[c] - m * a;
  }
}

// BN + ReLU. Optional outputs: swizzled f16 (next GEMM), linear f16 (pool /
// gather consumers), f32 in (b,c,n) layout (final network output).
__global__ void bn_relu_kernel(const float* __restrict__ Y, const float* __restrict__ scale,
                               const float* __restrict__ shift,
                               _Float16* __restrict__ Xsw, _Float16* __restrict__ Flin,
                               float* __restrict__ Fout,
                               int M, int Nn, int C, int total) {
  int i = blockIdx.x * blockDim.x + threadIdx.x;
  if (i >= total) return;
  int c = i / M; int m = i - c * M;
  float v = fmaxf(Y[i] * scale[c] + shift[c], 0.0f);
  if (Xsw) Xsw[swz_off(c, m, C >> 5)] = (_Float16)v;
  if (Flin) Flin[i] = (_Float16)v;
  if (Fout) {
    int b = m / Nn; int n = m - b * Nn;
    Fout[((size_t)b * C + c) * (size_t)Nn + n] = v;
  }
}

// Max over S samples: X[C][B*Np*S] -> out[C][B*Np]  (linear channel-major f16)
__global__ void maxpool_kernel(const _Float16* __restrict__ X, _Float16* __restrict__ out,
                               int S, int BNp, int total) {
  int i = blockIdx.x * blockDim.x + threadIdx.x;
  if (i >= total) return;
  int c = i / BNp; int m = i - c * BNp;
  const _Float16* src = X + ((size_t)c * BNp + m) * S;
  float best = -3.0e38f;
  for (int s = 0; s < S; ++s) best = fmaxf(best, (float)src[s]);
  out[i] = (_Float16)best;
}

// ---------------------------------------------------------------------------
// FP: 3-NN inverse-distance interpolation + concat -> swizzled Cin32 x (B*Nu).
// knf/unkf linear channel-major f16; unkf_xyz!=nullptr -> unk feats = raw xyz.
// ---------------------------------------------------------------------------
__global__ void fp_interp_kernel(const float* __restrict__ unk_xyz, const float* __restrict__ kn_xyz,
                                 const _Float16* __restrict__ knf, const _Float16* __restrict__ unkf,
                                 const float* __restrict__ unkf_xyz, _Float16* __restrict__ Xsw,
                                 int Nu, int Nk, int Ck, int Cu, int B, int Cin32, int total) {
  int t = blockIdx.x * blockDim.x + threadIdx.x;
  if (t >= total) return;
  int b = t / Nu; int n = t - b * Nu;
  const float* q = unk_xyz + (size_t)t * 3;
  const float* p = kn_xyz + (size_t)b * Nk * 3;
  float d0 = 3e38f, d1 = 3e38f, d2v = 3e38f;
  int i0 = 0, i1 = 0, i2 = 0;
  for (int i = 0; i < Nk; ++i) {
    float dx = p[i*3+0]-q[0], dy = p[i*3+1]-q[1], dz = p[i*3+2]-q[2];
    float d = dx*dx + dy*dy + dz*dz;
    if (d < d2v) {
      if (d < d1) {
        if (d < d0) { d2v = d1; i2 = i1; d1 = d0; i1 = i0; d0 = d; i0 = i; }
        else        { d2v = d1; i2 = i1; d1 = d;  i1 = i; }
      } else        { d2v = d;  i2 = i; }
    }
  }
  float w0 = 1.0f/(d0+1e-8f), w1 = 1.0f/(d1+1e-8f), w2 = 1.0f/(d2v+1e-8f);
  float wsum = w0 + w1 + w2; w0 /= wsum; w1 /= wsum; w2 /= wsum;
  int nkb = Cin32 >> 5;
  size_t BNk = (size_t)B * Nk;
  for (int c = 0; c < Ck; ++c) {
    const _Float16* f = knf + (size_t)c * BNk + (size_t)b * Nk;
    float v = w0*(float)f[i0] + w1*(float)f[i1] + w2*(float)f[i2];
    Xsw[swz_off(c, t, nkb)] = (_Float16)v;
  }
  if (unkf_xyz) {
    for (int c = 0; c < Cu; ++c)
      Xsw[swz_off(Ck + c, t, nkb)] = (_Float16)unkf_xyz[(size_t)t*3 + c];
  } else {
    size_t BNu = (size_t)B * Nu;
    for (int c = 0; c < Cu; ++c)
      Xsw[swz_off(Ck + c, t, nkb)] = unkf[(size_t)c * BNu + (size_t)b * Nu + n];
  }
  for (int c = Ck + Cu; c < Cin32; ++c)   // zero K-padding
    Xsw[swz_off(c, t, nkb)] = (_Float16)0.0f;
}

// Convert W[Cout][Cin] f32 -> f16, K-padded to Cin32, swizzled fragment order.
__global__ void wswz_kernel(const float* __restrict__ W, _Float16* __restrict__ out,
                            int Cin, int Cin32, int total) {
  int i = blockIdx.x * blockDim.x + threadIdx.x;
  if (i >= total) return;
  int r = i / Cin32, k = i - r * Cin32;
  float v = (k < Cin) ? W[(size_t)r * Cin + k] : 0.0f;
  out[swz_off(k, r, Cin32 >> 5)] = (_Float16)v;
}

// ---------------------------------------------------------------------------
extern "C" void kernel_launch(void* const* d_in, const int* in_sizes, int n_in,
                              void* d_out, int out_size, void* d_ws, size_t ws_size,
                              hipStream_t stream) {
  (void)in_sizes; (void)n_in; (void)out_size; (void)ws_size;
  const int B = 8, N0 = 4096, S = 20;
  const float* pc = (const float*)d_in[0];

  char* ws = (char*)d_ws;
  size_t off = 0;
  auto alloc = [&](size_t bytes) -> void* {
    void* p = (void*)(ws + off);
    off += WS_ALIGN(bytes);
    return p;
  };
  const size_t MAXE = (size_t)64 * 163840;  // largest operand: 64 x 163840
  _Float16* Xa    = (_Float16*)alloc(MAXE * sizeof(_Float16));  // swizzled ping
  _Float16* Xb    = (_Float16*)alloc(MAXE * sizeof(_Float16));  // swizzled pong / linear last
  float*    Yb    = (float*)alloc(MAXE * sizeof(float));
  float*    scale = (float*)alloc(512 * sizeof(float));
  float*    shift = (float*)alloc(512 * sizeof(float));
  float*    mind  = (float*)alloc((size_t)B * N0 * sizeof(float));
  int*      cidx1 = (int*)alloc((size_t)B*1024*sizeof(int));
  float*    nx1   = (float*)alloc((size_t)B*1024*3*sizeof(float));
  int*      nidx1 = (int*)alloc((size_t)B*1024*S*sizeof(int));
  int*      cidx2 = (int*)alloc((size_t)B*128*sizeof(int));
  float*    nx2   = (float*)alloc((size_t)B*128*3*sizeof(float));
  int*      nidx2 = (int*)alloc((size_t)B*128*S*sizeof(int));
  int*      cidx3 = (int*)alloc((size_t)B*16*sizeof(int));
  float*    nx3   = (float*)alloc((size_t)B*16*3*sizeof(float));
  int*      nidx3 = (int*)alloc((size_t)B*16*S*sizeof(int));
  _Float16* feats1 = (_Float16*)alloc((size_t)64*B*1024*sizeof(_Float16));
  _Float16* feats2 = (_Float16*)alloc((size_t)256*B*128*sizeof(_Float16));
  _Float16* feats3 = (_Float16*)alloc((size_t)512*B*16*sizeof(_Float16));
  _Float16* fpo2   = (_Float16*)alloc((size_t)256*B*128*sizeof(_Float16));
  _Float16* fpo1   = (_Float16*)alloc((size_t)256*B*1024*sizeof(_Float16));
  _Float16* wbuf   = (_Float16*)alloc((size_t)1048576*sizeof(_Float16));

  // Leaf order: pointcloud, then params['sa'] (3 stages x 3 layers x (W,g,b)),
  // then params['fp'] (3 stages).
  struct L { int widx; int cout; int cin; };
  const L sa1[3]  = {{1,32,6},{4,32,32},{7,64,32}};
  const L sa2[3]  = {{10,64,67},{13,64,64},{16,256,64}};
  const L sa3[3]  = {{19,256,259},{22,256,256},{25,512,256}};
  const L fp0l[3] = {{28,256,259},{31,256,256},{34,256,256}};
  const L fp1l[2] = {{37,256,320},{40,256,256}};
  const L fp2l[2] = {{43,256,768},{46,256,256}};

  auto cin32 = [](int c) { return (c + 31) & ~31; };

  // Convert + swizzle all weights to f16 (K-padded).
  _Float16* wptr[16];
  {
    const L* all[6] = {sa1, sa2, sa3, fp0l, fp1l, fp2l};
    const int nls[6] = {3,3,3,3,2,2};
    size_t woff = 0; int li = 0;
    for (int g = 0; g < 6; ++g) {
      for (int l = 0; l < nls[g]; ++l) {
        const L& LL = all[g][l];
        int k32 = cin32(LL.cin);
        int n = LL.cout * k32;
        wptr[li] = wbuf + woff;
        wswz_kernel<<<dim3((n+255)/256), dim3(256), 0, stream>>>(
            (const float*)d_in[LL.widx], wptr[li], LL.cin, k32, n);
        woff += (size_t)n;
        ++li;
      }
    }
  }

  // Shared MLP runner: swizzled f16 activations ping-pong Xa/Xb, f32 GEMM in Yb.
  auto run_mlp = [&](const L* Ls, int nL, int wbase, int M, int Nn,
                     _Float16* f16_final, float* f32_final) -> _Float16* {
    _Float16* bufs[2] = {Xa, Xb};
    int cur = 0;
    _Float16* last = Xa;
    for (int l = 0; l < nL; ++l) {
      const L& LL = Ls[l];
      int nkb = cin32(LL.cin) >> 5;
      if (LL.cout % 64 == 0) {
        int tiles = (LL.cout / 64) * (M / 16);
        wmma_gemm_rt<4><<<dim3(tiles), dim3(32), 0, stream>>>(
            wptr[wbase + l], bufs[cur], Yb, nkb, M);
      } else {
        int tiles = (LL.cout / 32) * (M / 16);
        wmma_gemm_rt<2><<<dim3(tiles), dim3(32), 0, stream>>>(
            wptr[wbase + l], bufs[cur], Yb, nkb, M);
      }
      bn_stats_kernel<<<dim3(LL.cout), dim3(256), 0, stream>>>(
          Yb, (const float*)d_in[LL.widx+1], (const float*)d_in[LL.widx+2],
          scale, shift, M);
      bool isLast = (l == nL - 1);
      _Float16* xsw  = isLast ? nullptr : bufs[cur ^ 1];
      _Float16* flin = isLast ? (f16_final ? f16_final : bufs[cur ^ 1]) : nullptr;
      float* fo = isLast ? f32_final : nullptr;
      int total = LL.cout * M;
      bn_relu_kernel<<<dim3((total+255)/256), dim3(256), 0, stream>>>(
          Yb, scale, shift, xsw, flin, fo, M, Nn, LL.cout, total);
      last = isLast ? flin : xsw;
      cur ^= 1;
    }
    return last;
  };

  // ---------------- SA1: 4096 -> 1024 centers ----------------
  {
    const int N = N0, Np = 1024, C32 = 32;
    fps_kernel<<<dim3(B), dim3(256), 0, stream>>>(pc, cidx1, mind, N, Np);
    gather_xyz<<<dim3((B*Np+255)/256), dim3(256), 0, stream>>>(pc, cidx1, nx1, N, Np, B*Np);
    knn_kernel<<<dim3((B*Np+63)/64), dim3(64), 0, stream>>>(pc, nx1, nidx1, N, Np, B*Np);
    int M = B * Np * S;  // 163840
    group_kernel<<<dim3((M+255)/256), dim3(256), 0, stream>>>(
        pc, nx1, nidx1, nullptr, Xa, N, Np, S, 3, B, C32, M);
    _Float16* act = run_mlp(sa1, 3, 0, M, 1, nullptr, nullptr);
    maxpool_kernel<<<dim3((64*B*Np+255)/256), dim3(256), 0, stream>>>(
        act, feats1, S, B*Np, 64*B*Np);
  }
  // ---------------- SA2: 1024 -> 128 centers ----------------
  {
    const int N = 1024, Np = 128, C32 = 96;  // 67 -> 96
    fps_kernel<<<dim3(B), dim3(256), 0, stream>>>(nx1, cidx2, mind, N, Np);
    gather_xyz<<<dim3((B*Np+255)/256), dim3(256), 0, stream>>>(nx1, cidx2, nx2, N, Np, B*Np);
    knn_kernel<<<dim3((B*Np+63)/64), dim3(64), 0, stream>>>(nx1, nx2, nidx2, N, Np, B*Np);
    int M = B * Np * S;  // 20480
    group_kernel<<<dim3((M+255)/256), dim3(256), 0, stream>>>(
        nx1, nx2, nidx2, feats1, Xa, N, Np, S, 64, B, C32, M);
    _Float16* act = run_mlp(sa2, 3, 3, M, 1, nullptr, nullptr);
    maxpool_kernel<<<dim3((256*B*Np+255)/256), dim3(256), 0, stream>>>(
        act, feats2, S, B*Np, 256*B*Np);
  }
  // ---------------- SA3: 128 -> 16 centers ----------------
  {
    const int N = 128, Np = 16, C32 = 288;  // 259 -> 288
    fps_kernel<<<dim3(B), dim3(256), 0, stream>>>(nx2, cidx3, mind, N, Np);
    gather_xyz<<<dim3((B*Np+255)/256), dim3(256), 0, stream>>>(nx2, cidx3, nx3, N, Np, B*Np);
    knn_kernel<<<dim3((B*Np+63)/64), dim3(64), 0, stream>>>(nx2, nx3, nidx3, N, Np, B*Np);
    int M = B * Np * S;  // 2560
    group_kernel<<<dim3((M+255)/256), dim3(256), 0, stream>>>(
        nx2, nx3, nidx3, feats2, Xa, N, Np, S, 256, B, C32, M);
    _Float16* act = run_mlp(sa3, 3, 6, M, 1, nullptr, nullptr);
    maxpool_kernel<<<dim3((512*B*Np+255)/256), dim3(256), 0, stream>>>(
        act, feats3, S, B*Np, 512*B*Np);
  }
  // ---------------- FP i=2: known=16, unknown=128 ----------------
  {
    const int Nu = 128, Nk = 16, Ck = 512, Cu = 256, C32 = 768;
    int M = B * Nu;  // 1024
    fp_interp_kernel<<<dim3((M+255)/256), dim3(256), 0, stream>>>(
        nx2, nx3, feats3, feats2, nullptr, Xa, Nu, Nk, Ck, Cu, B, C32, M);
    run_mlp(fp2l, 2, 14, M, 1, fpo2, nullptr);
  }
  // ---------------- FP i=1: known=128, unknown=1024 ----------------
  {
    const int Nu = 1024, Nk = 128, Ck = 256, Cu = 64, C32 = 320;
    int M = B * Nu;  // 8192
    fp_interp_kernel<<<dim3((M+255)/256), dim3(256), 0, stream>>>(
        nx1, nx2, fpo2, feats1, nullptr, Xa, Nu, Nk, Ck, Cu, B, C32, M);
    run_mlp(fp1l, 2, 12, M, 1, fpo1, nullptr);
  }
  // ---------------- FP i=0: known=1024, unknown=4096 -> d_out ----------------
  {
    const int Nu = 4096, Nk = 1024, Ck = 256, Cu = 3, C32 = 288;  // 259 -> 288
    int M = B * Nu;  // 32768
    fp_interp_kernel<<<dim3((M+255)/256), dim3(256), 0, stream>>>(
        pc, nx1, fpo1, nullptr, pc, Xa, Nu, Nk, Ck, Cu, B, C32, M);
    run_mlp(fp0l, 3, 9, M, Nu, nullptr, (float*)d_out);
  }
}
